// SpectralInteraction_59785944760359
// MI455X (gfx1250) — matrix operand order repaired and checked by reading, not compiled, not run on hardware
//
#include <hip/hip_runtime.h>
#include <math.h>

// ---------------- CDNA5 WMMA / TDM types ----------------
typedef __bf16 bf16_t;
typedef __attribute__((ext_vector_type(16))) __bf16 v16bf;
typedef __attribute__((ext_vector_type(8)))  float  v8f;
typedef __attribute__((ext_vector_type(4)))  unsigned int v4u;
typedef __attribute__((ext_vector_type(4)))  int v4i;
typedef __attribute__((ext_vector_type(8)))  int v8i;

#define WMMA_BF16(a, b, c) \
  __builtin_amdgcn_wmma_f32_16x16x32_bf16(false, (a), false, (b), (short)0, (c), false, false)

#if defined(__gfx1250__) && __has_builtin(__builtin_amdgcn_tensor_load_to_lds)
#define HAVE_TDM 1
#else
#define HAVE_TDM 0
#endif

#if HAVE_TDM
// 2D tile DMA: global (bf16, row stride = strideElems) -> LDS, rows x cols elements.
// D# layout per CDNA5 ISA 8.3/8.4 (group0: count/lds/global/type, group1: dims/strides).
__device__ inline void tdm_load_tile_2d(unsigned int lds_addr, const bf16_t* gptr,
                                        unsigned int cols, unsigned int rows,
                                        unsigned int strideElems) {
  unsigned long long ga = (unsigned long long)(const void*)gptr;
  v4u g0;
  g0[0] = 1u;                                            // count=1 (valid), user fields 0
  g0[1] = lds_addr;                                      // lds_addr[31:0]
  g0[2] = (unsigned int)(ga & 0xFFFFFFFFu);              // global_addr[31:0]
  g0[3] = (unsigned int)((ga >> 32) & 0x01FFFFFFu) | (2u << 30);  // addr[56:32] | type=2
  v8i g1;
  g1[0] = (1 << 16);                                     // data_size=1 (2 bytes)
  g1[1] = (int)((cols & 0xFFFFu) << 16);                 // tensor_dim0[15:0]
  g1[2] = (int)(((cols >> 16) & 0xFFFFu) | ((rows & 0xFFFFu) << 16)); // dim0 hi | tensor_dim1 lo
  g1[3] = (int)((cols & 0xFFFFu) << 16);                 // tile_dim0 = cols
  g1[4] = (int)(rows & 0xFFFFu);                         // tile_dim1 = rows, tile_dim2 = 0
  g1[5] = (int)strideElems;                              // tensor_dim0_stride[31:0]
  g1[6] = 0;
  g1[7] = 0;
  v4i z4 = {};
#if __has_include(<hip/amd_detail/amd_gfx1250_TDM.h>)
  v8i z8 = {};
  __builtin_amdgcn_tensor_load_to_lds(g0, g1, z4, z4, z8, 0);   // clang-23 / therock 6-arg form
#else
  __builtin_amdgcn_tensor_load_to_lds(g0, g1, z4, z4, 0);       // ROCm 7.2 5-arg form
#endif
}
#endif

// A-matrix 16x32 bf16 fragment from row-major [16][ldk] storage.
// lane<16 holds M=lane, K=0..15; lane>=16 holds M=lane-16, K=16..31 (contiguous, matches B fragments)
__device__ inline v16bf load_afrag(const bf16_t* __restrict__ base, int ldk, int lane) {
  int m  = lane & 15;
  int kb = (lane & 16) ? 16 : 0;
  const bf16_t* p = base + m * ldk;
  v16bf a;
#pragma unroll
  for (int j = 0; j < 8; ++j)  a[j] = p[kb + j];
#pragma unroll
  for (int j = 8; j < 16; ++j) a[j] = p[kb + j];   // K = kb+8 .. kb+15
  return a;
}

// B-matrix 32x16 bf16 fragment where memory holds Bt[n][k] row-major (B transposed):
// per lane the 16 K-values are contiguous: lane<16 -> K=0..15, lane>=16 -> K=16..31.
__device__ inline v16bf load_bfrag_T(const bf16_t* __restrict__ base, int ldk, int lane) {
  int n  = lane & 15;
  int kb = (lane & 16) ? 16 : 0;
  return *(const v16bf*)(base + n * ldk + kb);
}

// ---------------- constants ----------------
#define NB   4
#define NL   1024
#define NH   8
#define NE   64
#define NF   513          // rfft bins
#define NFP  528          // NF padded to 33 * 16
#define NFP2 544          // NF padded to 17 * 32 (K-dim of irfft GEMM)
#define BH   (NB * NH)    // 32 heads
#define BC1_ELEMS (33 * 32 * 512)   // [ntile(freq)][kstep(s)][lane*16+j]
#define BC2_ELEMS (64 * 17 * 512)   // [ntile(n)][kstep(freq)][lane*16+j]

// ---------------- kernel 0: pre-swizzled cos/sin B-fragment tables ----------------
__global__ void init_tables_kernel(bf16_t* __restrict__ BC1, bf16_t* __restrict__ BS1,
                                   bf16_t* __restrict__ BC2, bf16_t* __restrict__ BS2,
                                   float* __restrict__ acc) {
  int id = blockIdx.x * blockDim.x + threadIdx.x;
  const float w = 6.283185307179586f / 1024.0f;
  if (id < BC1_ELEMS) {   // GEMM K-dim = s (time), N-dim = freq
    int j = id & 15, lane = (id >> 4) & 31, rest = id >> 9;
    int ks = rest & 31, nt = rest >> 5;
    int s  = ks * 32 + ((lane & 16) ? 16 : 0) + j;
    int kf = nt * 16 + (lane & 15);
    float c = 0.f, sn = 0.f;
    if (kf < NF) { int ph = (s * kf) & 1023; float a = ph * w; c = cosf(a); sn = sinf(a); }
    BC1[id] = (bf16_t)c; BS1[id] = (bf16_t)sn;
  }
  if (id < BC2_ELEMS) {   // GEMM K-dim = freq, N-dim = n (time)
    int j = id & 15, lane = (id >> 4) & 31, rest = id >> 9;
    int ks = rest % 17, nt = rest / 17;
    int kf = ks * 32 + ((lane & 16) ? 16 : 0) + j;
    int n  = nt * 16 + (lane & 15);
    float c = 0.f, sn = 0.f;
    if (kf < NF) { int ph = (kf * n) & 1023; float a = ph * w; c = cosf(a); sn = sinf(a); }
    BC2[id] = (bf16_t)c; BS2[id] = (bf16_t)sn;
  }
  if (id < BH * NF) acc[id] = 0.f;
}

// ---------------- kernel 1: q/|q|, k/|k| -> bf16, [bh][l][e] layout ----------------
__global__ void normalize_kernel(const float* __restrict__ Q, const float* __restrict__ K,
                                 bf16_t* __restrict__ Qn, bf16_t* __restrict__ Kn) {
  int v = blockIdx.x * blockDim.x + threadIdx.x;
  if (v >= 2 * BH * NL) return;
  int isK = v >> 15, vec = v & 32767;
  int b = vec >> 13, h = (vec >> 10) & 7, l = vec & 1023;
  const float* src = (isK ? K : Q) + (((size_t)b * NL + l) * NH + h) * NE;
  bf16_t* dst = (isK ? Kn : Qn) + (size_t)vec * NE;
  float ss = 0.f;
#pragma unroll 8
  for (int e = 0; e < NE; ++e) { float x = src[e]; ss += x * x; }
  float inv = rsqrtf(fmaxf(ss, 1e-24f));
#pragma unroll 8
  for (int e = 0; e < NE; ++e) dst[e] = (bf16_t)(src[e] * inv);
}

// ---------------- kernel 1b: values -> bf16 [bh][e][n] (TDM-friendly row-major) ----------------
__global__ void transpose_v_kernel(const float* __restrict__ V, bf16_t* __restrict__ Vn) {
  int id = blockIdx.x * blockDim.x + threadIdx.x;
  if (id >= BH * NE * NL) return;
  int n = id & 1023, e = (id >> 10) & 63, bh = id >> 16;
  int b = bh >> 3, h = bh & 7;
  Vn[id] = (bf16_t)V[(((size_t)b * NL + n) * NH + h) * NE + e];
}

// ---------------- kernel 2: S=exp(QK^T) -> F=S@cos -> rowwise softmax -> acc ----------------
__global__ void __launch_bounds__(32)
spectral_sim_kernel(const bf16_t* __restrict__ Qn, const bf16_t* __restrict__ Kn,
                    const bf16_t* __restrict__ BC1, float* __restrict__ acc) {
  extern __shared__ char smem[];
  float*  Fsh = (float*)smem;                       // 16 x 528 f32
  bf16_t* Ssh = (bf16_t*)(smem + 16 * NFP * 4);     // 16 x 1024 bf16
  __shared__ float mxs[16], den[16];

  int lane = threadIdx.x;
  int mt = blockIdx.x & 63, bh = blockIdx.x >> 6;
  int Mrow = (lane & 16) ? 8 : 0, Ncol = lane & 15;

  const bf16_t* Qbase = Qn + ((size_t)bh * NL + mt * 16) * NE;
  v16bf aq0 = load_afrag(Qbase,      NE, lane);
  v16bf aq1 = load_afrag(Qbase + 32, NE, lane);

  // S row-slab (16 x 1024) = exp(cosine-similarity), kept in LDS
  for (int st = 0; st < 64; ++st) {
    const bf16_t* Kbase = Kn + ((size_t)bh * NL + st * 16) * NE;
    v16bf b0 = load_bfrag_T(Kbase,      NE, lane);
    v16bf b1 = load_bfrag_T(Kbase + 32, NE, lane);
    v8f c = {};
    c = WMMA_BF16(aq0, b0, c);
    c = WMMA_BF16(aq1, b1, c);
#pragma unroll
    for (int i = 0; i < 8; ++i)
      Ssh[(i + Mrow) * NL + st * 16 + Ncol] = (bf16_t)__expf(c[i]);
  }
  __syncthreads();

  // F (16 x 528) = S @ cos-basis  (the 34-GFLOP GEMM)
  for (int nt = 0; nt < 33; ++nt) {
    const bf16_t* bt = BC1 + (size_t)nt * 32 * 512;
    v8f c = {};
#pragma unroll 4
    for (int ks = 0; ks < 32; ++ks) {
      v16bf a = load_afrag(Ssh + ks * 32, NL, lane);
      v16bf b = *(const v16bf*)(bt + ks * 512 + lane * 16);
      if (ks + 8 < 32) __builtin_prefetch(bt + (ks + 8) * 512 + lane * 16, 0, 1);
      c = WMMA_BF16(a, b, c);
    }
#pragma unroll
    for (int i = 0; i < 8; ++i)
      Fsh[(i + Mrow) * NFP + nt * 16 + Ncol] = c[i];
  }
  __syncthreads();

  // per-row softmax(scale * F) and sum over the 16 rows -> global accumulator
  const float scale = 0.125f;   // 1/sqrt(64)
  if (lane < 16) {
    float mx = -1e30f;
    for (int k = 0; k < NF; ++k) mx = fmaxf(mx, Fsh[lane * NFP + k]);
    float s = 0.f;
    for (int k = 0; k < NF; ++k) s += __expf((Fsh[lane * NFP + k] - mx) * scale);
    mxs[lane] = mx; den[lane] = 1.0f / s;
  }
  __syncthreads();
  for (int k = lane; k < NF; k += 32) {
    float t = 0.f;
#pragma unroll
    for (int r = 0; r < 16; ++r)
      t += __expf((Fsh[r * NFP + k] - mxs[r]) * scale) * den[r];
    atomicAdd(&acc[bh * NF + k], t);
  }
}

// ---------------- kernel 3: A = softmax(acc) per head ----------------
__global__ void softmax_acc_kernel(const float* __restrict__ acc, float* __restrict__ Aw) {
  __shared__ float red[64];
  int bh = blockIdx.x, t = threadIdx.x;
  const float* a = acc + bh * NF;
  float mx = -1e30f;
  for (int k = t; k < NF; k += 64) mx = fmaxf(mx, a[k]);
  red[t] = mx; __syncthreads();
  for (int s = 32; s > 0; s >>= 1) { if (t < s) red[t] = fmaxf(red[t], red[t + s]); __syncthreads(); }
  mx = red[0]; __syncthreads();
  float sm = 0.f;
  for (int k = t; k < NF; k += 64) sm += __expf(a[k] - mx);
  red[t] = sm; __syncthreads();
  for (int s = 32; s > 0; s >>= 1) { if (t < s) red[t] += red[t + s]; __syncthreads(); }
  float inv = 1.0f / red[0];
  for (int k = t; k < NF; k += 64) Aw[bh * NF + k] = __expf(a[k] - mx) * inv;
}

// ---------------- kernel 4: rfft(v) -> scale real by A -> irfft -> transpose store ----------------
__global__ void __launch_bounds__(128)
spectral_filter_kernel(const bf16_t* __restrict__ Vn, const float* __restrict__ Aw,
                       const bf16_t* __restrict__ BC1, const bf16_t* __restrict__ BS1,
                       const bf16_t* __restrict__ BC2, const bf16_t* __restrict__ BS2,
                       float* __restrict__ out) {
  extern __shared__ char smem[];
  bf16_t* vsh  = (bf16_t*)smem;            // 16 x 1024
  bf16_t* rsh  = vsh + 16 * NL;            // 16 x 544 (A[k]*w_k/N * Re)
  bf16_t* ish  = rsh + 16 * NFP2;          // 16 x 544 (w_k/N * (-Im))
  float*  awsh = (float*)(ish + 16 * NFP2);
  float*  wsh  = awsh + NFP2;

  int tid = threadIdx.x, lane = tid & 31, wv = tid >> 5;
  int et = blockIdx.x & 3, bh = blockIdx.x >> 2;
  int b = bh >> 3, h = bh & 7;
  int Mrow = (lane & 16) ? 8 : 0, Ncol = lane & 15;

  const bf16_t* vtile = Vn + ((size_t)bh * NE + et * 16) * NL;
#if HAVE_TDM
  if (wv == 0) {
    // Tensor Data Mover: DMA the 16x1024 bf16 slab straight into LDS (TENSORcnt-tracked).
    tdm_load_tile_2d((unsigned int)(unsigned long long)(void*)vsh, vtile, NL, 16, NL);
  }
#else
  for (int idx = tid; idx < 16 * NL; idx += 128) vsh[idx] = vtile[idx];
#endif
  for (int idx = tid; idx < 16 * NFP2; idx += 128) { rsh[idx] = (bf16_t)0.f; ish[idx] = (bf16_t)0.f; }
  for (int k = tid; k < NFP2; k += 128) {
    float wk = ((k == 0) || (k == 512)) ? (1.f / 1024.f) : (2.f / 1024.f);
    wsh[k]  = (k < NF) ? wk : 0.f;
    awsh[k] = (k < NF) ? wk * Aw[bh * NF + k] : 0.f;
  }
#if HAVE_TDM
  __builtin_amdgcn_s_wait_tensorcnt(0);
#endif
  __syncthreads();

  // forward DFT of v: Vr[k]=sum v cos, Vi[k]=sum v sin  (Vi = -Imag)
  for (int kt = wv; kt < 33; kt += 4) {
    const bf16_t* bc = BC1 + (size_t)kt * 32 * 512;
    const bf16_t* bs = BS1 + (size_t)kt * 32 * 512;
    v8f cr = {}, ci = {};
#pragma unroll 4
    for (int ks = 0; ks < 32; ++ks) {
      v16bf a  = load_afrag(vsh + ks * 32, NL, lane);
      v16bf bC = *(const v16bf*)(bc + ks * 512 + lane * 16);
      v16bf bS = *(const v16bf*)(bs + ks * 512 + lane * 16);
      cr = WMMA_BF16(a, bC, cr);
      ci = WMMA_BF16(a, bS, ci);
    }
#pragma unroll
    for (int i = 0; i < 8; ++i) {
      int k = kt * 16 + Ncol, M = i + Mrow;
      rsh[M * NFP2 + k] = (bf16_t)(cr[i] * awsh[k]);
      ish[M * NFP2 + k] = (bf16_t)(ci[i] * wsh[k]);
    }
  }
  __syncthreads();

  // irfft: out[e][n] = sum_k ( rsh[k]*cos(2pi k n/N) + ish[k]*sin(2pi k n/N) )
  for (int nt = wv; nt < 64; nt += 4) {
    const bf16_t* bc = BC2 + (size_t)nt * 17 * 512;
    const bf16_t* bs = BS2 + (size_t)nt * 17 * 512;
    v8f c = {};
#pragma unroll 4
    for (int ks = 0; ks < 17; ++ks) {
      v16bf ar = load_afrag(rsh + ks * 32, NFP2, lane);
      v16bf ai = load_afrag(ish + ks * 32, NFP2, lane);
      v16bf bC = *(const v16bf*)(bc + ks * 512 + lane * 16);
      v16bf bS = *(const v16bf*)(bs + ks * 512 + lane * 16);
      c = WMMA_BF16(ar, bC, c);
      c = WMMA_BF16(ai, bS, c);
    }
#pragma unroll
    for (int i = 0; i < 8; ++i) {
      int e = et * 16 + i + Mrow;
      int n = nt * 16 + Ncol;
      out[(((size_t)b * NL + n) * NE + e) * NH + h] = c[i];   // (B,L,E,H)
    }
  }
}

// ---------------- launcher ----------------
extern "C" void kernel_launch(void* const* d_in, const int* in_sizes, int n_in,
                              void* d_out, int out_size, void* d_ws, size_t ws_size,
                              hipStream_t stream) {
  const float* q = (const float*)d_in[0];
  const float* k = (const float*)d_in[1];
  const float* v = (const float*)d_in[2];
  float* out = (float*)d_out;

  char* ws = (char*)d_ws;
  size_t off = 0;
  auto carve = [&](size_t bytes) -> char* {
    char* p = ws + off;
    off = (off + bytes + 255) & ~(size_t)255;
    return p;
  };
  bf16_t* Qn  = (bf16_t*)carve((size_t)BH * NL * NE * 2);
  bf16_t* Kn  = (bf16_t*)carve((size_t)BH * NL * NE * 2);
  bf16_t* Vn  = (bf16_t*)carve((size_t)BH * NE * NL * 2);
  bf16_t* BC1 = (bf16_t*)carve((size_t)BC1_ELEMS * 2);
  bf16_t* BS1 = (bf16_t*)carve((size_t)BC1_ELEMS * 2);
  bf16_t* BC2 = (bf16_t*)carve((size_t)BC2_ELEMS * 2);
  bf16_t* BS2 = (bf16_t*)carve((size_t)BC2_ELEMS * 2);
  float*  acc = (float*)carve((size_t)BH * NF * 4);
  float*  Aw  = (float*)carve((size_t)BH * NF * 4);

  init_tables_kernel<<<(BC2_ELEMS + 255) / 256, 256, 0, stream>>>(BC1, BS1, BC2, BS2, acc);
  normalize_kernel<<<(2 * BH * NL + 255) / 256, 256, 0, stream>>>(q, k, Qn, Kn);
  transpose_v_kernel<<<(BH * NE * NL + 255) / 256, 256, 0, stream>>>(v, Vn);

  size_t shm2 = (size_t)16 * NFP * 4 + (size_t)16 * NL * 2;            // ~66 KB (320 KB LDS/WGP)
  spectral_sim_kernel<<<BH * (NL / 16), 32, shm2, stream>>>(Qn, Kn, BC1, acc);

  softmax_acc_kernel<<<BH, 64, 0, stream>>>(acc, Aw);

  size_t shm4 = (size_t)16 * NL * 2 + 2 * (size_t)16 * NFP2 * 2 + 2 * (size_t)NFP2 * 4; // ~72 KB
  spectral_filter_kernel<<<BH * 4, 128, shm4, stream>>>(Vn, Aw, BC1, BS1, BC2, BS2, out);
}
